// simplified_attention_47699906790048
// MI455X (gfx1250) — compile-verified, matrix operand order checked by reading, not run on hardware
//
#include <hip/hip_runtime.h>
#include <hip/hip_bf16.h>
#include <math.h>

// ---------------------------------------------------------------------------
// Problem constants (from reference)
// ---------------------------------------------------------------------------
#define B_      4
#define S_      1024
#define HSIZE_  2048
#define NH_     32
#define NKV_    8
#define HD_     64          // head dim
#define BS_     (B_ * S_)   // 4096 token rows

typedef __attribute__((ext_vector_type(16))) __bf16 bf16x16;
typedef __attribute__((ext_vector_type(8)))  float  floatx8;
typedef unsigned int v4u __attribute__((ext_vector_type(4)));
typedef int          v8i __attribute__((ext_vector_type(8)));
typedef int          v4i __attribute__((ext_vector_type(4)));

union Frag16 {
    bf16x16      v;
    unsigned int u[8];
    uint4        q[2];
};

__device__ __forceinline__ floatx8 zero8() {
    floatx8 z = {0.f, 0.f, 0.f, 0.f, 0.f, 0.f, 0.f, 0.f};
    return z;
}

__device__ __forceinline__ floatx8 wmma_bf16(const bf16x16& a, const bf16x16& b,
                                             const floatx8& c) {
    return __builtin_amdgcn_wmma_f32_16x16x32_bf16(
        /*neg_a=*/false, a, /*neg_b=*/false, b,
        /*c_mod=*/(short)0, c, /*reuse_a=*/false, /*reuse_b=*/false);
}

// Fragment k-offset pattern (16-bit, ISA 7.12.2):
//   lanes 0-15 (half=0): k pairs start at {0,2,4,6,16,18,20,22}
//   lanes 16-31 (half=1): same + 8
__device__ __forceinline__ int frag_koff(int j) { return (j < 4) ? (2 * j) : (8 + 2 * j); }

// A fragment: 16x32 tile of row-major bf16, rows m0..m0+15, cols k0..k0+31.
__device__ __forceinline__ void load_a_bf16(Frag16& f, const __bf16* __restrict__ A,
                                            int m0, int k0, int ld, int lane) {
    const int half = lane >> 4;
    const int m    = m0 + (lane & 15);
    const __bf16* row = A + (size_t)m * ld + k0 + half * 8;
#pragma unroll
    for (int j = 0; j < 8; ++j) {
        f.u[j] = *(const unsigned int*)(row + frag_koff(j));   // fuses to b128s
    }
}

// A fragment sourced from an f32 row-major matrix (convert to bf16 in regs).
__device__ __forceinline__ void load_a_f32(Frag16& f, const float* __restrict__ A,
                                           int m0, int k0, int ld, int lane) {
    const int half = lane >> 4;
    const int m    = m0 + (lane & 15);
    const float* row = A + (size_t)m * ld + k0 + half * 8;
#pragma unroll
    for (int j = 0; j < 8; ++j) {
        const int k = frag_koff(j);
        f.v[2 * j]     = (__bf16)row[k];
        f.v[2 * j + 1] = (__bf16)row[k + 1];
    }
}

// B fragment from an LDS tile stored K-contiguous per column: lds[n*pitch + k].
// pitch in elements, multiple of 8 so the two 16B chunks stay aligned.
template <int PITCH>
__device__ __forceinline__ void load_b_lds(Frag16& f, const __bf16* lds,
                                           int n0, int k0, int lane) {
    const int half = lane >> 4;
    const int n    = n0 + (lane & 15);
    const uint4* p = (const uint4*)(lds + (size_t)n * PITCH + k0 + half * 8);
    f.q[0] = p[0];   // k offsets half*8 + 0..7
    f.q[1] = p[2];   // k offsets half*8 + 16..23 (byte offset +32)
}

// C/D 16x16 f32 tile store: VGPR r holds (M = m0 + r + 8*half, N = n0 + lane&15)
template <typename OutT>
__device__ __forceinline__ void store_c(const floatx8& c, OutT* __restrict__ C,
                                        int m0, int n0, int ldc, int lane) {
    const int half = lane >> 4;
    const int n    = n0 + (lane & 15);
#pragma unroll
    for (int r = 0; r < 8; ++r) {
        C[(size_t)(m0 + r + 8 * half) * ldc + n] = (OutT)c[r];
    }
}

// ---------------------------------------------------------------------------
// TDM helper: DMA a rows x cols (bf16, row-major) tile from global memory into
// LDS offset 0 (dynamic shared memory), packed contiguous.
// D# layout per CDNA5 ISA ch.8 (group0 128b, group1 256b; groups 2/3 zero for
// 2D tensors). This toolchain's builtin takes 6 args (extra int32x8 group).
// ---------------------------------------------------------------------------
__device__ __forceinline__ void tdm_load_tile_2d(const __bf16* gsrc,
                                                 unsigned rows, unsigned cols,
                                                 unsigned row_stride_elems) {
    const unsigned long long ga = (unsigned long long)(uintptr_t)gsrc;
    v4u g0;
    g0[0] = 1u;                               // count=1, user descriptor
    g0[1] = 0u;                               // lds_addr = 0 (dynamic LDS base)
    g0[2] = (unsigned)(ga & 0xFFFFFFFFu);     // global_addr[31:0]
    g0[3] = (unsigned)((ga >> 32) & 0x1FFFFFFu) | (2u << 30);  // addr[56:32] | type=2

    v8i g1;
    g1[0] = (int)(1u << 16);                  // wg_mask=0, data_size=1 (2 bytes)
    g1[1] = (int)(cols << 16);                // tensor_dim0[15:0]
    g1[2] = (int)(rows << 16);                // tensor_dim0[31:16]=0 | tensor_dim1[15:0]
    g1[3] = (int)(cols << 16);                // tensor_dim1[31:16]=0 | tile_dim0
    g1[4] = (int)rows;                        // tile_dim1 | tile_dim2=0
    g1[5] = (int)row_stride_elems;            // tensor_dim0_stride[31:0]
    g1[6] = 0;                                // stride hi | tensor_dim1_stride lo
    g1[7] = 0;

    v4i gz4 = {0, 0, 0, 0};
    v8i gz8 = {0, 0, 0, 0, 0, 0, 0, 0};
    __builtin_amdgcn_tensor_load_to_lds(g0, g1, gz4, gz4, gz8, 0);
}

// ---------------------------------------------------------------------------
// f32 -> bf16 conversion
// ---------------------------------------------------------------------------
__global__ void cvt_f32_bf16_ker(const float* __restrict__ src, __bf16* __restrict__ dst, int n) {
    int i = blockIdx.x * blockDim.x + threadIdx.x;
    if (i < n) dst[i] = (__bf16)src[i];
}

// ---------------------------------------------------------------------------
// Tiled bf16 WMMA GEMM with LDS-staged weight tiles (transposed on write).
// C[M,N] = A[M,K] * W[K,N]; 256 threads = 8 waves; block tile 128x64.
// ---------------------------------------------------------------------------
#define WPITCH 48   // 96B row pitch: 16B-aligned chunks, spreads LDS banks

template <typename OutT>
__global__ __launch_bounds__(256) void gemm_bf16_ker(const __bf16* __restrict__ A,
                                                     const __bf16* __restrict__ W,
                                                     OutT* __restrict__ C,
                                                     int M, int N, int K) {
    __shared__ __align__(16) __bf16 ldsW[64 * WPITCH];

    const int lane = threadIdx.x & 31;
    const int wave = threadIdx.x >> 5;
    const int wr = wave >> 1;
    const int wc = wave & 1;
    const int m0   = blockIdx.y * 128 + wr * 32;
    const int nblk = blockIdx.x * 64;
    const int n0w  = wc * 32;

    floatx8 acc00 = zero8(), acc01 = zero8(), acc10 = zero8(), acc11 = zero8();

    const int sk = (threadIdx.x * 8) >> 6;   // 0..31 (k within tile)
    const int sn = (threadIdx.x * 8) & 63;   // 0..56 step 8 (n within tile)

    for (int k0 = 0; k0 < K; k0 += 32) {
        // ---- cooperative stage: W[k0+sk][nblk+sn .. +7] -> ldsW[n][k] ----
        {
            uint4 d = *(const uint4*)(W + (size_t)(k0 + sk) * N + nblk + sn);
            const __bf16* e = (const __bf16*)&d;
#pragma unroll
            for (int j = 0; j < 8; ++j) ldsW[(sn + j) * WPITCH + sk] = e[j];
        }
        __syncthreads();

        __builtin_prefetch(A + (size_t)(m0 + (lane & 15)) * K + k0 + 64, 0, 0);

        Frag16 a0, a1, b0, b1;
        load_a_bf16(a0, A, m0,      k0, K, lane);
        load_a_bf16(a1, A, m0 + 16, k0, K, lane);
        load_b_lds<WPITCH>(b0, ldsW, n0w,      0, lane);
        load_b_lds<WPITCH>(b1, ldsW, n0w + 16, 0, lane);

        acc00 = wmma_bf16(a0.v, b0.v, acc00);
        acc01 = wmma_bf16(a0.v, b1.v, acc01);
        acc10 = wmma_bf16(a1.v, b0.v, acc10);
        acc11 = wmma_bf16(a1.v, b1.v, acc11);
        __syncthreads();
    }

    store_c(acc00, C, m0,      nblk + n0w,      N, lane);
    store_c(acc01, C, m0,      nblk + n0w + 16, N, lane);
    store_c(acc10, C, m0 + 16, nblk + n0w,      N, lane);
    store_c(acc11, C, m0 + 16, nblk + n0w + 16, N, lane);
}

// ---------------------------------------------------------------------------
// RoPE (interleaved) + L2 norm, relayout [ (b*S+s), h*64+d ] -> [ (b*H+h)*S+s, d ]
// ---------------------------------------------------------------------------
__global__ void rope_norm_ker(const __bf16* __restrict__ proj,
                              const float* __restrict__ cosT,
                              const float* __restrict__ sinT,
                              __bf16* __restrict__ outN,
                              int nHeads, int projLd) {
    int idx = blockIdx.x * blockDim.x + threadIdx.x;
    int total = BS_ * nHeads;
    if (idx >= total) return;
    int h  = idx % nHeads;
    int bs = idx / nHeads;
    int s  = bs & (S_ - 1);
    int b  = bs >> 10;

    const __bf16* src = proj + (size_t)bs * projLd + h * HD_;
    const float* cp = cosT + (size_t)s * (HD_ / 2);
    const float* sp = sinT + (size_t)s * (HD_ / 2);

    float x[HD_];
    float ss = 0.f;
#pragma unroll
    for (int i = 0; i < HD_ / 2; ++i) {
        float xr = (float)src[2 * i];
        float xi = (float)src[2 * i + 1];
        float c = cp[i], sn = sp[i];
        float orr = xr * c - xi * sn;
        float oii = xr * sn + xi * c;
        x[2 * i]     = orr;
        x[2 * i + 1] = oii;
        ss += orr * orr + oii * oii;
    }
    float r = rsqrtf(ss * (1.0f / HD_) + 1e-6f);

    __bf16* dst = outN + ((size_t)(b * nHeads + h) * S_ + s) * HD_;
#pragma unroll
    for (int i = 0; i < HD_; ++i) dst[i] = (__bf16)(x[i] * r);
}

// V relayout only: [ (b*S+s), kv*64+d ] -> [ (b*KV+kv)*S+s, d ]
__global__ void v_relayout_ker(const __bf16* __restrict__ proj, __bf16* __restrict__ outN) {
    int idx = blockIdx.x * blockDim.x + threadIdx.x;
    int total = BS_ * NKV_;
    if (idx >= total) return;
    int h  = idx % NKV_;
    int bs = idx / NKV_;
    int s  = bs & (S_ - 1);
    int b  = bs >> 10;
    const __bf16* src = proj + (size_t)bs * (NKV_ * HD_) + h * HD_;
    __bf16* dst = outN + ((size_t)(b * NKV_ + h) * S_ + s) * HD_;
#pragma unroll
    for (int i = 0; i < HD_; ++i) dst[i] = src[i];
}

// ---------------------------------------------------------------------------
// Logits: attn[bh, q, k] = scale * (Qn . Kn) + causal mask.
// The 64-key x 64-dim K tile (8 KB) is DMA'd into LDS by the Tensor Data
// Mover (wave 0 issues TENSOR_LOAD_TO_LDS, waits on TENSORcnt, then the
// workgroup barrier releases the other waves). Fragments: 2x ds_load_b128.
// Dynamic shared memory => LDS byte address 0 in the descriptor.
// ---------------------------------------------------------------------------
#define KPITCH 64   // TDM writes the tile packed: pitch = tile_dim0 = 64

__global__ __launch_bounds__(256) void qk_logits_ker(const __bf16* __restrict__ Qn,
                                                     const __bf16* __restrict__ Kn,
                                                     float* __restrict__ attn) {
    extern __shared__ __align__(16) __bf16 ldsK[];   // 64*64 bf16 = 8 KB dynamic

    const int lane = threadIdx.x & 31;
    const int wave = threadIdx.x >> 5;
    const int wr = wave >> 1, wc = wave & 1;

    const int bh = blockIdx.z;
    const int b  = bh >> 5;
    const int h  = bh & 31;
    const int kv = h >> 2;

    const __bf16* qh = Qn + (size_t)bh * S_ * HD_;
    const __bf16* kh = Kn + (size_t)(b * NKV_ + kv) * S_ * HD_;
    float* out = attn + (size_t)bh * S_ * S_;

    const int q0   = blockIdx.y * 128 + wr * 32;
    const int nblk = blockIdx.x * 64;
    const int n0w  = wc * 32;

    // ---- TDM: DMA K tile (rows nblk..nblk+63, 64 dims) into LDS ----
    if (wave == 0) {
        tdm_load_tile_2d(kh + (size_t)nblk * HD_, 64, 64, HD_);
        __builtin_amdgcn_s_wait_tensorcnt(0);
    }
    __syncthreads();

    floatx8 acc00 = zero8(), acc01 = zero8(), acc10 = zero8(), acc11 = zero8();

#pragma unroll
    for (int k0 = 0; k0 < HD_; k0 += 32) {
        Frag16 a0, a1, b0, b1;
        load_a_bf16(a0, qh, q0,      k0, HD_, lane);
        load_a_bf16(a1, qh, q0 + 16, k0, HD_, lane);
        load_b_lds<KPITCH>(b0, ldsK, n0w,      k0, lane);
        load_b_lds<KPITCH>(b1, ldsK, n0w + 16, k0, lane);
        acc00 = wmma_bf16(a0.v, b0.v, acc00);
        acc01 = wmma_bf16(a0.v, b1.v, acc01);
        acc10 = wmma_bf16(a1.v, b0.v, acc10);
        acc11 = wmma_bf16(a1.v, b1.v, acc11);
    }

    const float scale = 0.125f;
    const int half = lane >> 4;
    const int nl   = lane & 15;
#pragma unroll
    for (int ti = 0; ti < 2; ++ti) {
        const floatx8& cA = ti ? acc10 : acc00;
        const floatx8& cB = ti ? acc11 : acc01;
#pragma unroll
        for (int r = 0; r < 8; ++r) {
            int m  = q0 + ti * 16 + r + 8 * half;
            int nA = nblk + n0w + nl;
            int nB = nblk + n0w + 16 + nl;
            out[(size_t)m * S_ + nA] = cA[r] * scale + ((nA <= m) ? 0.f : -1.0e9f);
            out[(size_t)m * S_ + nB] = cB[r] * scale + ((nB <= m) ? 0.f : -1.0e9f);
        }
    }
}

// ---------------------------------------------------------------------------
// Softmax in place: one wave per 1024-element row.
// ---------------------------------------------------------------------------
__global__ __launch_bounds__(256) void softmax_ker(float* __restrict__ attn) {
    const int lane = threadIdx.x & 31;
    const int wave = threadIdx.x >> 5;
    const size_t row = (size_t)blockIdx.x * 8 + wave;
    float* p = attn + row * S_;

    float v[32];
    float mx = -3.0e38f;
#pragma unroll
    for (int i = 0; i < 32; ++i) {
        v[i] = p[i * 32 + lane];
        mx = fmaxf(mx, v[i]);
    }
#pragma unroll
    for (int off = 16; off >= 1; off >>= 1) mx = fmaxf(mx, __shfl_xor(mx, off, 32));

    float sum = 0.f;
#pragma unroll
    for (int i = 0; i < 32; ++i) {
        v[i] = __expf(v[i] - mx);
        sum += v[i];
    }
#pragma unroll
    for (int off = 16; off >= 1; off >>= 1) sum += __shfl_xor(sum, off, 32);

    const float inv = 1.0f / sum;
#pragma unroll
    for (int i = 0; i < 32; ++i) p[i * 32 + lane] = v[i] * inv;
}

// ---------------------------------------------------------------------------
// ctx = attn (f32 -> bf16 in regs) @ V (bf16), V tile LDS-staged per k-step.
// ---------------------------------------------------------------------------
__global__ __launch_bounds__(256) void av_ker(const float* __restrict__ attn,
                                              const __bf16* __restrict__ Vn,
                                              __bf16* __restrict__ ctx) {
    __shared__ __align__(16) __bf16 ldsV[64 * WPITCH];

    const int lane = threadIdx.x & 31;
    const int wave = threadIdx.x >> 5;

    const int bh = blockIdx.z;
    const int b  = bh >> 5;
    const int h  = bh & 31;
    const int kv = h >> 2;

    const float*  ah = attn + (size_t)bh * S_ * S_;
    const __bf16* vh = Vn + (size_t)(b * NKV_ + kv) * S_ * HD_;

    const int q0 = blockIdx.y * 128 + wave * 16;

    floatx8 acc[4] = {zero8(), zero8(), zero8(), zero8()};

    const int sk = (threadIdx.x * 8) >> 6;
    const int sn = (threadIdx.x * 8) & 63;

    for (int k0 = 0; k0 < S_; k0 += 32) {
        {
            uint4 d = *(const uint4*)(vh + (size_t)(k0 + sk) * HD_ + sn);
            const __bf16* e = (const __bf16*)&d;
#pragma unroll
            for (int j = 0; j < 8; ++j) ldsV[(sn + j) * WPITCH + sk] = e[j];
        }
        __syncthreads();

        Frag16 a;
        load_a_f32(a, ah, q0, k0, S_, lane);
#pragma unroll
        for (int j = 0; j < 4; ++j) {
            Frag16 bj;
            load_b_lds<WPITCH>(bj, ldsV, j * 16, 0, lane);
            acc[j] = wmma_bf16(a.v, bj.v, acc[j]);
        }
        __syncthreads();
    }

    const int half = lane >> 4;
    const int nl   = lane & 15;
#pragma unroll
    for (int j = 0; j < 4; ++j) {
#pragma unroll
        for (int r = 0; r < 8; ++r) {
            int m   = q0 + r + 8 * half;
            int col = h * HD_ + j * 16 + nl;
            ctx[((size_t)b * S_ + m) * HSIZE_ + col] = (__bf16)acc[j][r];
        }
    }
}

// ---------------------------------------------------------------------------
// Host-side launcher
// ---------------------------------------------------------------------------
extern "C" void kernel_launch(void* const* d_in, const int* in_sizes, int n_in,
                              void* d_out, int out_size, void* d_ws, size_t ws_size,
                              hipStream_t stream) {
    (void)in_sizes; (void)n_in; (void)out_size; (void)ws_size;

    const float* hs  = (const float*)d_in[0];
    const float* Wq  = (const float*)d_in[3];
    const float* Wk  = (const float*)d_in[4];
    const float* Wv  = (const float*)d_in[5];
    const float* Wo  = (const float*)d_in[6];
    const float* cosT = (const float*)d_in[7];
    const float* sinT = (const float*)d_in[8];

    float* outF = (float*)d_out;
    float* attn = (float*)d_out + (size_t)BS_ * HSIZE_;

    char* ws = (char*)d_ws;
    size_t off = 0;
    auto carve = [&](size_t bytes) { char* p = ws + off; off += (bytes + 255) & ~(size_t)255; return p; };

    __bf16* hs_bf = (__bf16*)carve((size_t)BS_ * HSIZE_ * 2);
    __bf16* wq_bf = (__bf16*)carve((size_t)HSIZE_ * HSIZE_ * 2);
    __bf16* wk_bf = (__bf16*)carve((size_t)HSIZE_ * NKV_ * HD_ * 2);
    __bf16* wv_bf = (__bf16*)carve((size_t)HSIZE_ * NKV_ * HD_ * 2);
    __bf16* wo_bf = (__bf16*)carve((size_t)HSIZE_ * HSIZE_ * 2);
    __bf16* qproj = (__bf16*)carve((size_t)BS_ * HSIZE_ * 2);
    __bf16* kproj = (__bf16*)carve((size_t)BS_ * NKV_ * HD_ * 2);
    __bf16* vproj = (__bf16*)carve((size_t)BS_ * NKV_ * HD_ * 2);
    __bf16* qn    = (__bf16*)carve((size_t)BS_ * HSIZE_ * 2);
    __bf16* kn    = (__bf16*)carve((size_t)BS_ * NKV_ * HD_ * 2);
    __bf16* vn    = (__bf16*)carve((size_t)BS_ * NKV_ * HD_ * 2);
    __bf16* ctx   = (__bf16*)carve((size_t)BS_ * HSIZE_ * 2);

    const int T = 256;
    auto blocks = [](size_t n, int t) { return (unsigned)((n + t - 1) / t); };

    cvt_f32_bf16_ker<<<blocks((size_t)BS_ * HSIZE_, T), T, 0, stream>>>(hs, hs_bf, BS_ * HSIZE_);
    cvt_f32_bf16_ker<<<blocks((size_t)HSIZE_ * HSIZE_, T), T, 0, stream>>>(Wq, wq_bf, HSIZE_ * HSIZE_);
    cvt_f32_bf16_ker<<<blocks((size_t)HSIZE_ * NKV_ * HD_, T), T, 0, stream>>>(Wk, wk_bf, HSIZE_ * NKV_ * HD_);
    cvt_f32_bf16_ker<<<blocks((size_t)HSIZE_ * NKV_ * HD_, T), T, 0, stream>>>(Wv, wv_bf, HSIZE_ * NKV_ * HD_);
    cvt_f32_bf16_ker<<<blocks((size_t)HSIZE_ * HSIZE_, T), T, 0, stream>>>(Wo, wo_bf, HSIZE_ * HSIZE_);

    gemm_bf16_ker<__bf16><<<dim3(HSIZE_ / 64, BS_ / 128), T, 0, stream>>>(
        hs_bf, wq_bf, qproj, BS_, HSIZE_, HSIZE_);
    gemm_bf16_ker<__bf16><<<dim3((NKV_ * HD_) / 64, BS_ / 128), T, 0, stream>>>(
        hs_bf, wk_bf, kproj, BS_, NKV_ * HD_, HSIZE_);
    gemm_bf16_ker<__bf16><<<dim3((NKV_ * HD_) / 64, BS_ / 128), T, 0, stream>>>(
        hs_bf, wv_bf, vproj, BS_, NKV_ * HD_, HSIZE_);

    rope_norm_ker<<<blocks((size_t)BS_ * NH_, T), T, 0, stream>>>(qproj, cosT, sinT, qn, NH_, NH_ * HD_);
    rope_norm_ker<<<blocks((size_t)BS_ * NKV_, T), T, 0, stream>>>(kproj, cosT, sinT, kn, NKV_, NKV_ * HD_);
    v_relayout_ker<<<blocks((size_t)BS_ * NKV_, T), T, 0, stream>>>(vproj, vn);

    // dynamic LDS: 64x64 bf16 K tile = 8192 bytes, DMA'd by the TDM
    qk_logits_ker<<<dim3(S_ / 64, S_ / 128, B_ * NH_), T, 64 * 64 * 2, stream>>>(qn, kn, attn);
    softmax_ker<<<blocks((size_t)B_ * NH_ * S_, 8), T, 0, stream>>>(attn);
    av_ker<<<dim3(1, S_ / 128, B_ * NH_), T, 0, stream>>>(attn, vn, ctx);

    gemm_bf16_ker<float><<<dim3(HSIZE_ / 64, BS_ / 128), T, 0, stream>>>(
        ctx, wo_bf, outF, BS_, HSIZE_, HSIZE_);
}
